// EdgeNet_33423435497580
// MI455X (gfx1250) — matrix operand order repaired and checked by reading, not compiled, not run on hardware
//
#include <hip/hip_runtime.h>
#include <math.h>

#define NN   1024
#define INF_ 128
#define H0_  192
#define H1_  96

typedef __attribute__((ext_vector_type(16))) _Float16 v16h;
typedef __attribute__((ext_vector_type(8)))  _Float16 v8h;
typedef __attribute__((ext_vector_type(8)))  float    v8f;

// ---------------- WMMA fragment loaders (16x16x32 f16, wave32) ----------------
// A (16x32, MxK): lane L<16 -> row M=L, K 0..15
//                 lane L>=16 -> row M=L-16, K 16..31  (16 contiguous halves).
__device__ __forceinline__ v16h load_frag_a(const _Float16* base, int row0, int stride, int k0) {
  const int lane = threadIdx.x & 31;
  const _Float16* p = base + (row0 + (lane & 15)) * stride + (k0 + ((lane >> 4) << 4));
  union { v16h v; v8h h[2]; } u;
  u.h[0] = *(const v8h*)(p);
  u.h[1] = *(const v8h*)(p + 8);
  return u.v;
}
// B (32x16, KxN) stored row-per-output-channel [n][k]: lane L<16 -> col N=L, K 0..15
//                 lane L>=16 -> col N=L-16, K 16..31  (16 contiguous halves).
__device__ __forceinline__ v16h load_frag_b(const _Float16* base, int col0, int stride, int k0) {
  const int lane = threadIdx.x & 31;
  const _Float16* p = base + (col0 + (lane & 15)) * stride + (k0 + ((lane >> 4) << 4));
  union { v16h v; v8h h[2]; } u;
  u.h[0] = *(const v8h*)(p);
  u.h[1] = *(const v8h*)(p + 8);
  return u.v;
}

// ---------------- prep: fold BN into f16 weights, convert features ----------------
__global__ __launch_bounds__(256) void prep_kernel(
    const float* __restrict__ nf,
    const float* __restrict__ W0, const float* __restrict__ g0, const float* __restrict__ b0,
    const float* __restrict__ m0, const float* __restrict__ v0,
    const float* __restrict__ W1, const float* __restrict__ g1, const float* __restrict__ b1,
    const float* __restrict__ m1, const float* __restrict__ v1,
    const float* __restrict__ Wout,
    _Float16* __restrict__ W0h, _Float16* __restrict__ W1h, _Float16* __restrict__ BoutP,
    float* __restrict__ t0, float* __restrict__ t1, _Float16* __restrict__ nfh)
{
  const int idx = blockIdx.x * 256 + threadIdx.x;
  if (idx < NN * INF_) nfh[idx] = (_Float16)nf[idx];
  if (idx < H0_ * INF_) {
    const int c = idx >> 7;
    const float s = g0[c] * rsqrtf(v0[c] + 1e-5f);
    W0h[idx] = (_Float16)(W0[idx] * s);
  }
  if (idx < H1_ * H0_) {
    const int d = idx / H0_;
    const float s = g1[d] * rsqrtf(v1[d] + 1e-5f);
    W1h[idx] = (_Float16)(W1[idx] * s);
  }
  if (idx < 16 * 104) {
    const int r = idx / 104, k = idx % 104;
    BoutP[idx] = (r == 0 && k < H1_) ? (_Float16)Wout[k] : (_Float16)0.0f;
  }
  if (idx < H0_) { const float s = g0[idx] * rsqrtf(v0[idx] + 1e-5f); t0[idx] = b0[idx] - m0[idx] * s; }
  if (idx < H1_) { const float s = g1[idx] * rsqrtf(v1[idx] + 1e-5f); t1[idx] = b1[idx] - m1[idx] * s; }
}

// ---------------- main fused kernel: one 8x8 (i,j) tile per workgroup ----------------
__global__ __launch_bounds__(128) void edgenet_main(
    const _Float16* __restrict__ nfh,
    const _Float16* __restrict__ W0h, const float* __restrict__ t0,
    const _Float16* __restrict__ W1h, const float* __restrict__ t1,
    const _Float16* __restrict__ BoutP, const float* __restrict__ boutg,
    float* __restrict__ simout)
{
  __shared__ _Float16 sA [64 * 136];   // |a_i - a_j|, 64 pairs x 128 (padded)
  __shared__ _Float16 sH [64 * 200];   // layer-1 activations, 64 x 192 (padded)
  __shared__ _Float16 sH2[64 * 104];   // layer-2 activations, 64 x 96  (padded)

  const int tid  = threadIdx.x;
  const int lane = tid & 31;
  const int wave = tid >> 5;
  const int i0 = blockIdx.y << 3;
  const int j0 = blockIdx.x << 3;

  // ---- stage A: pairwise abs-diff into LDS (f16) ----
  {
    const int p  = tid >> 1;            // pair 0..63
    const int kb = (tid & 1) << 6;      // 0 or 64
    const int gi = i0 + (p >> 3);
    const int gj = j0 + (p & 7);
    const _Float16* ai = nfh + gi * INF_ + kb;
    const _Float16* aj = nfh + gj * INF_ + kb;
    _Float16* dst = sA + p * 136 + kb;
    #pragma unroll
    for (int k = 0; k < 64; k += 8) {
      v8h x = *(const v8h*)(ai + k);
      v8h y = *(const v8h*)(aj + k);
      v8h d;
      #pragma unroll
      for (int e = 0; e < 8; ++e) {
        _Float16 df = x[e] - y[e];
        d[e] = df < (_Float16)0 ? (_Float16)(-df) : df;
      }
      *(v8h*)(dst + k) = d;
    }
  }
  __syncthreads();

  // ---- GEMM1: [64x128] x [128x192] -> BN+leaky -> sH (f16) ----
  for (int T = wave; T < 48; T += 4) {          // 4x12 output tiles of 16x16
    const int mt = T & 3;
    const int nt = T >> 2;
    v8f acc = {};
    #pragma unroll
    for (int kc = 0; kc < 4; ++kc) {
      v16h a = load_frag_a(sA, mt << 4, 136, kc << 5);
      v16h b = load_frag_b(W0h, nt << 4, INF_, kc << 5);
      acc = __builtin_amdgcn_wmma_f32_16x16x32_f16(false, a, false, b, (short)0, acc, false, false);
    }
    const int col = (nt << 4) + (lane & 15);
    const float tb = t0[col];
    _Float16* drow = sH + ((mt << 4) + ((lane >> 4) << 3)) * 200 + col;
    #pragma unroll
    for (int r = 0; r < 8; ++r) {
      float y = acc[r] + tb;
      y = y >= 0.0f ? y : 0.01f * y;
      drow[r * 200] = (_Float16)y;
    }
  }
  __syncthreads();

  // ---- GEMM2: [64x192] x [192x96] -> BN+leaky -> sH2 (f16) ----
  for (int T = wave; T < 24; T += 4) {          // 4x6 output tiles
    const int mt = T & 3;
    const int nt = T >> 2;
    v8f acc = {};
    #pragma unroll
    for (int kc = 0; kc < 6; ++kc) {
      v16h a = load_frag_a(sH, mt << 4, 200, kc << 5);
      v16h b = load_frag_b(W1h, nt << 4, H0_, kc << 5);
      acc = __builtin_amdgcn_wmma_f32_16x16x32_f16(false, a, false, b, (short)0, acc, false, false);
    }
    const int col = (nt << 4) + (lane & 15);
    const float tb = t1[col];
    _Float16* drow = sH2 + ((mt << 4) + ((lane >> 4) << 3)) * 104 + col;
    #pragma unroll
    for (int r = 0; r < 8; ++r) {
      float y = acc[r] + tb;
      y = y >= 0.0f ? y : 0.01f * y;
      drow[r * 104] = (_Float16)y;
    }
  }
  __syncthreads();

  // ---- GEMM3: [64x96] x [96x16] (col 0 = Wout) -> sigmoid -> sim ----
  {
    const int mt = wave;                        // one 16-row tile per wave
    v8f acc = {};
    #pragma unroll
    for (int kc = 0; kc < 3; ++kc) {
      v16h a = load_frag_a(sH2, mt << 4, 104, kc << 5);
      v16h b = load_frag_b(BoutP, 0, 104, kc << 5);
      acc = __builtin_amdgcn_wmma_f32_16x16x32_f16(false, a, false, b, (short)0, acc, false, false);
    }
    if ((lane & 15) == 0) {                     // lanes 0 and 16 hold column N=0
      const float bo = boutg[0];
      const int rbase = (mt << 4) + ((lane >> 4) << 3);
      #pragma unroll
      for (int r = 0; r < 8; ++r) {
        const int p = rbase + r;
        const float logit = acc[r] + bo;
        const float sim = 1.0f / (1.0f + __expf(-logit));
        simout[(i0 + (p >> 3)) * NN + (j0 + (p & 7))] = sim;
      }
    }
  }
}

// ---------------- column sums (fixed order -> deterministic) ----------------
__global__ __launch_bounds__(256) void colsum_kernel(const float* __restrict__ sim,
                                                     float* __restrict__ colsum)
{
  const int j = blockIdx.x * 256 + threadIdx.x;
  float s = 0.0f;
  for (int i = 0; i < NN; ++i) s += sim[i * NN + j] + 1e-6f;
  colsum[j] = s + 1.0f;   // the single i==j identity term
}

__global__ __launch_bounds__(256) void edge_kernel(const float* __restrict__ sim,
                                                   const float* __restrict__ colsum,
                                                   float* __restrict__ edge)
{
  const int idx = blockIdx.x * 256 + threadIdx.x;
  const int i = idx >> 10, j = idx & (NN - 1);
  const float v = sim[idx] + (i == j ? 1.0f : 0.0f) + 1e-6f;
  edge[idx] = v / colsum[j];
}

// ---------------- launch ----------------
extern "C" void kernel_launch(void* const* d_in, const int* in_sizes, int n_in,
                              void* d_out, int out_size, void* d_ws, size_t ws_size,
                              hipStream_t stream) {
  (void)in_sizes; (void)n_in; (void)out_size; (void)ws_size;
  const float* nf   = (const float*)d_in[0];
  const float* W0   = (const float*)d_in[1];
  const float* g0   = (const float*)d_in[2];
  const float* b0   = (const float*)d_in[3];
  const float* m0   = (const float*)d_in[4];
  const float* v0   = (const float*)d_in[5];
  const float* W1   = (const float*)d_in[6];
  const float* g1   = (const float*)d_in[7];
  const float* b1   = (const float*)d_in[8];
  const float* m1   = (const float*)d_in[9];
  const float* v1   = (const float*)d_in[10];
  const float* Wout = (const float*)d_in[11];
  const float* bout = (const float*)d_in[12];

  char* ws = (char*)d_ws;
  _Float16* W0h   = (_Float16*)(ws + 0);        // 192*128*2 = 49152
  _Float16* W1h   = (_Float16*)(ws + 49152);    // 96*192*2  = 36864
  _Float16* BoutP = (_Float16*)(ws + 86016);    // 16*104*2  = 3328
  float*    t0    = (float*)   (ws + 89600);    // 192*4
  float*    t1    = (float*)   (ws + 90368);    // 96*4
  _Float16* nfh   = (_Float16*)(ws + 90880);    // 1024*128*2 = 262144
  float*    colsum= (float*)   (ws + 353024);   // 1024*4

  float* edge = (float*)d_out;            // [1, N, N]
  float* sim  = (float*)d_out + NN * NN;  // [N, N]

  prep_kernel<<<512, 256, 0, stream>>>(nf, W0, g0, b0, m0, v0,
                                       W1, g1, b1, m1, v1, Wout,
                                       W0h, W1h, BoutP, t0, t1, nfh);
  edgenet_main<<<dim3(NN / 8, NN / 8), 128, 0, stream>>>(nfh, W0h, t0, W1h, t1,
                                                         BoutP, bout, sim);
  colsum_kernel<<<NN / 256, 256, 0, stream>>>(sim, colsum);
  edge_kernel<<<(NN * NN) / 256, 256, 0, stream>>>(sim, colsum, edge);
}
